// Seq2Seq_LSTM_46557445489130
// MI455X (gfx1250) — matrix operand order repaired and checked by reading, not compiled, not run on hardware
//
#include <hip/hip_runtime.h>
#include <math.h>

// ---------------- types ----------------
typedef __attribute__((ext_vector_type(16))) __bf16 v16bf;
typedef __attribute__((ext_vector_type(8)))  float  v8f;

union V16 { uint4 q[2]; v16bf v; };

#define B_  256
#define T_  96
#define H_  512
#define L_  5
#define PRED_ 14
#define G4H 2048   // 4*H

// ---------------- helpers ----------------
__device__ __forceinline__ unsigned short f2b(float f) {
    unsigned int u = __float_as_uint(f);
    u += 0x7FFFu + ((u >> 16) & 1u);          // round-to-nearest-even
    return (unsigned short)(u >> 16);
}
__device__ __forceinline__ float b2f(unsigned short h) {
    return __uint_as_float(((unsigned int)h) << 16);
}
__device__ __forceinline__ float sigm(float x) { return 1.0f / (1.0f + __expf(-x)); }

// K=512 bf16 GEMM accumulate into 4 16x16 f32 tiles (one gate slice per wave).
// A: h row (bf16, ld=512). W: row-major (4H x 512) bf16; B(k,n)=W[n,k].
__device__ __forceinline__ void gemm_k512(const unsigned short* __restrict__ Arow,
                                          const unsigned short* __restrict__ W,
                                          int nBase, int r, int half, v8f acc[4]) {
    const unsigned short* wr0 = W + (size_t)(nBase +  0 + r) * H_ + half * 16;
    const unsigned short* wr1 = W + (size_t)(nBase + 16 + r) * H_ + half * 16;
    const unsigned short* wr2 = W + (size_t)(nBase + 32 + r) * H_ + half * 16;
    const unsigned short* wr3 = W + (size_t)(nBase + 48 + r) * H_ + half * 16;
    const unsigned short* ar  = Arow + half * 16;
#pragma unroll 4
    for (int kk = 0; kk < H_; kk += 32) {
        V16 a;
        a.q[0] = *(const uint4*)(ar + kk);        // K = kk + half*16 .. +7
        a.q[1] = *(const uint4*)(ar + kk + 8);    // K = kk + half*16 + 8 .. +15
        V16 b0, b1, b2, b3;
        b0.q[0] = *(const uint4*)(wr0 + kk); b0.q[1] = *(const uint4*)(wr0 + kk + 8);
        b1.q[0] = *(const uint4*)(wr1 + kk); b1.q[1] = *(const uint4*)(wr1 + kk + 8);
        b2.q[0] = *(const uint4*)(wr2 + kk); b2.q[1] = *(const uint4*)(wr2 + kk + 8);
        b3.q[0] = *(const uint4*)(wr3 + kk); b3.q[1] = *(const uint4*)(wr3 + kk + 8);
        acc[0] = __builtin_amdgcn_wmma_f32_16x16x32_bf16(false, a.v, false, b0.v, (short)0, acc[0], false, false);
        acc[1] = __builtin_amdgcn_wmma_f32_16x16x32_bf16(false, a.v, false, b1.v, (short)0, acc[1], false, false);
        acc[2] = __builtin_amdgcn_wmma_f32_16x16x32_bf16(false, a.v, false, b2.v, (short)0, acc[2], false, false);
        acc[3] = __builtin_amdgcn_wmma_f32_16x16x32_bf16(false, a.v, false, b3.v, (short)0, acc[3], false, false);
    }
}

// ---------------- fused LSTM cell step ----------------
// grid = 128 blocks (16 row-blocks x 8 col-blocks), block = 128 threads (4 waves).
// Wave w computes gate w's 16x64 slice: gates = h_in@Whh^T [+ x2@W2^T] + bias [+ xs[m]*Ws[n]].
// Then LDS exchange + fused cell update: c = sig(f)*c + sig(i)*tanh(g); h = sig(o)*tanh(c).
__global__ void lstm_step_kernel(const unsigned short* __restrict__ h_in,   // B x H bf16
                                 float* __restrict__ c,                     // B x H f32 (in-place)
                                 const unsigned short* __restrict__ Whh,    // 4H x H bf16
                                 const float* __restrict__ bias,            // 4H
                                 const unsigned short* __restrict__ x2,     // B x H bf16 or null
                                 const unsigned short* __restrict__ W2,     // 4H x H bf16 or null
                                 const float* __restrict__ xs, int xsStride, int xsOff,
                                 const float* __restrict__ Ws,              // 4H or null
                                 unsigned short* __restrict__ h_out,        // B x H bf16
                                 unsigned short* __restrict__ seq_out) {    // B x H bf16 or null
    __shared__ float lds[4 * 16 * 64];   // [gate][row16][col64] = 16 KB

    const int rb   = blockIdx.x >> 3;          // 0..15 row block
    const int cb   = blockIdx.x & 7;           // 0..7  col block
    const int w    = threadIdx.x >> 5;         // gate 0..3 (i,f,g,o)
    const int lane = threadIdx.x & 31;
    const int half = lane >> 4;
    const int r    = lane & 15;
    const int nBase = w * H_ + cb * 64;        // gate-column base in 4H

    v8f acc[4] = { v8f{}, v8f{}, v8f{}, v8f{} };

    const unsigned short* Arow = h_in + (size_t)(rb * 16 + r) * H_;
    gemm_k512(Arow, Whh, nBase, r, half, acc);
    if (x2) {
        const unsigned short* Xrow = x2 + (size_t)(rb * 16 + r) * H_;
        gemm_k512(Xrow, W2, nBase, r, half, acc);
    }

    // scalar-input rank-1 term (IN==1 layers): xs[m] * Ws[n]
    float xv[8];
    if (xs) {
#pragma unroll
        for (int j = 0; j < 8; ++j)
            xv[j] = xs[(size_t)(rb * 16 + j + half * 8) * xsStride + xsOff];
    }

    // scatter gate values (C/D layout: lane%16 = N, vgpr j = M = j + half*8) to LDS
#pragma unroll
    for (int t = 0; t < 4; ++t) {
        const int nloc = t * 16 + r;           // 0..63 within block cols
        const int n    = nBase + nloc;         // global gate column
        const float bv = bias[n];
        const float wv = xs ? Ws[n] : 0.0f;
#pragma unroll
        for (int j = 0; j < 8; ++j) {
            const int lm = j + half * 8;       // local row 0..15
            float v = acc[t][j] + bv;
            if (xs) v += xv[j] * wv;
            lds[(w * 16 + lm) * 64 + nloc] = v;
        }
    }
    __syncthreads();

    // fused cell update over the 16x64 hidden slice
    for (int e = threadIdx.x; e < 16 * 64; e += 128) {
        const int lm = e >> 6;
        const int jc = e & 63;
        const float iv = sigm(lds[(0 * 16 + lm) * 64 + jc]);
        const float fv = sigm(lds[(1 * 16 + lm) * 64 + jc]);
        const float gv = tanhf(lds[(2 * 16 + lm) * 64 + jc]);
        const float ov = sigm(lds[(3 * 16 + lm) * 64 + jc]);
        const size_t idx = (size_t)(rb * 16 + lm) * H_ + cb * 64 + jc;
        const float cn = fv * c[idx] + iv * gv;
        c[idx] = cn;
        const unsigned short hb = f2b(ov * tanhf(cn));
        h_out[idx] = hb;
        if (seq_out) seq_out[idx] = hb;
    }
}

// ---------------- small kernels ----------------
__global__ void conv_bf16_kernel(const float* __restrict__ in, unsigned short* __restrict__ out, int n) {
    for (int i = blockIdx.x * blockDim.x + threadIdx.x; i < n; i += gridDim.x * blockDim.x)
        out[i] = f2b(in[i]);
}
__global__ void zero_f32_kernel(float* __restrict__ p, int n) {
    for (int i = blockIdx.x * blockDim.x + threadIdx.x; i < n; i += gridDim.x * blockDim.x)
        p[i] = 0.0f;
}
__global__ void x0_kernel(const float* __restrict__ x, float* __restrict__ ybuf) {
    int b = blockIdx.x * blockDim.x + threadIdx.x;
    if (b < B_) ybuf[b] = x[(size_t)b * T_ + (T_ - 1)];
}
// y = h4 @ fcW^T + fcb ; write d_out[b*PRED + step] and feed back ybuf
__global__ void fc_kernel(const unsigned short* __restrict__ h4, const float* __restrict__ fcW,
                          const float* __restrict__ fcb, float* __restrict__ out, int step,
                          float* __restrict__ ybuf) {
    int b = blockIdx.x * blockDim.x + threadIdx.x;
    if (b >= B_) return;
    float s = 0.0f;
    const unsigned short* hr = h4 + (size_t)b * H_;
    for (int j = 0; j < H_; ++j) s += b2f(hr[j]) * fcW[j];
    s += fcb[0];
    out[(size_t)b * PRED_ + step] = s;
    ybuf[b] = s;
}

// ---------------- workspace layout (bytes) ----------------
#define N_WIH   (4u * G4H * H_)     // 4,194,304 elems
#define N_WHH   (5u * G4H * H_)     // 5,242,880 elems
#define OFF_ENCWIH  ((size_t)0)
#define OFF_ENCWHH  (OFF_ENCWIH + (size_t)N_WIH * 2)
#define OFF_DECWIH  (OFF_ENCWHH + (size_t)N_WHH * 2)
#define OFF_DECWHH  (OFF_DECWIH + (size_t)N_WIH * 2)
#define OFF_HBUF    (OFF_DECWHH + (size_t)N_WHH * 2)                 // 5 layers x 2 parities x B*H bf16
#define OFF_CBUF    (OFF_HBUF + (size_t)L_ * 2 * B_ * H_ * 2)        // 5 layers x B*H f32
#define OFF_SEQA    (OFF_CBUF + (size_t)L_ * B_ * H_ * 4)            // T*B*H bf16
#define OFF_SEQB    (OFF_SEQA + (size_t)T_ * B_ * H_ * 2)
#define OFF_YBUF    (OFF_SEQB + (size_t)T_ * B_ * H_ * 2)            // B f32

extern "C" void kernel_launch(void* const* d_in, const int* in_sizes, int n_in,
                              void* d_out, int out_size, void* d_ws, size_t ws_size,
                              hipStream_t stream) {
    const float* x       = (const float*)d_in[0];
    const float* encWih0 = (const float*)d_in[1];
    const float* encWih  = (const float*)d_in[2];
    const float* encWhh  = (const float*)d_in[3];
    const float* encB    = (const float*)d_in[4];
    const float* decWih0 = (const float*)d_in[5];
    const float* decWih  = (const float*)d_in[6];
    const float* decWhh  = (const float*)d_in[7];
    const float* decB    = (const float*)d_in[8];
    const float* fcW     = (const float*)d_in[9];
    const float* fcb     = (const float*)d_in[10];

    char* ws = (char*)d_ws;
    unsigned short* encWihB = (unsigned short*)(ws + OFF_ENCWIH);
    unsigned short* encWhhB = (unsigned short*)(ws + OFF_ENCWHH);
    unsigned short* decWihB = (unsigned short*)(ws + OFF_DECWIH);
    unsigned short* decWhhB = (unsigned short*)(ws + OFF_DECWHH);
    auto hbuf = [&](int l, int p) -> unsigned short* {
        return (unsigned short*)(ws + OFF_HBUF + ((size_t)(l * 2 + p) * B_ * H_) * 2);
    };
    auto cbuf = [&](int l) -> float* {
        return (float*)(ws + OFF_CBUF + (size_t)l * B_ * H_ * 4);
    };
    unsigned short* seqPair[2] = { (unsigned short*)(ws + OFF_SEQA),
                                   (unsigned short*)(ws + OFF_SEQB) };
    float* ybuf = (float*)(ws + OFF_YBUF);

    // 1) weights fp32 -> bf16 (L2-resident, ~38 MB)
    conv_bf16_kernel<<<1024, 256, 0, stream>>>(encWih, encWihB, (int)N_WIH);
    conv_bf16_kernel<<<1024, 256, 0, stream>>>(encWhh, encWhhB, (int)N_WHH);
    conv_bf16_kernel<<<1024, 256, 0, stream>>>(decWih, decWihB, (int)N_WIH);
    conv_bf16_kernel<<<1024, 256, 0, stream>>>(decWhh, decWhhB, (int)N_WHH);
    // 2) zero h ping-pong (bf16 zeros == f32 zeros bitwise) and c
    zero_f32_kernel<<<1024, 256, 0, stream>>>((float*)(ws + OFF_HBUF),
                                              (int)((OFF_SEQA - OFF_HBUF) / 4));

    // 3) encoder: per (layer, step) fused step kernel; recurrence via stream order
    for (int l = 0; l < L_; ++l) {
        const unsigned short* Whh = encWhhB + (size_t)l * G4H * H_;
        const float* bias = encB + (size_t)l * G4H;
        const int outIdx = l & 1;
        const int inIdx  = outIdx ^ 1;
        for (int t = 0; t < T_; ++t) {
            const unsigned short* x2 = (l == 0) ? nullptr
                : seqPair[inIdx] + (size_t)t * B_ * H_;
            const unsigned short* W2 = (l == 0) ? nullptr
                : encWihB + (size_t)(l - 1) * G4H * H_;
            const float* xs = (l == 0) ? x : nullptr;
            const float* Wsv = (l == 0) ? encWih0 : nullptr;
            unsigned short* so = (l < L_ - 1)
                ? seqPair[outIdx] + (size_t)t * B_ * H_ : nullptr;
            lstm_step_kernel<<<128, 128, 0, stream>>>(
                hbuf(l, t & 1), cbuf(l), Whh, bias, x2, W2,
                xs, T_, t, Wsv, hbuf(l, (t + 1) & 1), so);
        }
    }

    // 4) decoder: state buffers continue at parity T%2 == 0
    x0_kernel<<<2, 128, 0, stream>>>(x, ybuf);
    for (int s = 0; s < PRED_; ++s) {
        const int t = T_ + s;
        for (int l = 0; l < L_; ++l) {
            const unsigned short* Whh = decWhhB + (size_t)l * G4H * H_;
            const float* bias = decB + (size_t)l * G4H;
            const unsigned short* x2 = (l == 0) ? nullptr : hbuf(l - 1, (t + 1) & 1);
            const unsigned short* W2 = (l == 0) ? nullptr
                : decWihB + (size_t)(l - 1) * G4H * H_;
            const float* xs = (l == 0) ? ybuf : nullptr;
            const float* Wsv = (l == 0) ? decWih0 : nullptr;
            lstm_step_kernel<<<128, 128, 0, stream>>>(
                hbuf(l, t & 1), cbuf(l), Whh, bias, x2, W2,
                xs, 1, 0, Wsv, hbuf(l, (t + 1) & 1), nullptr);
        }
        fc_kernel<<<2, 128, 0, stream>>>(hbuf(L_ - 1, (t + 1) & 1), fcW, fcb,
                                         (float*)d_out, s, ybuf);
    }
}
